// HeteroGCN_pyg_17119739641951
// MI455X (gfx1250) — matrix operand order repaired and checked by reading, not compiled, not run on hardware
//
#include <hip/hip_runtime.h>
#include <hip/hip_bf16.h>

// ---- sizes from the reference ----
#define NP    50000
#define EPPI  600000
#define NPAIR 100000
#define FDIM  128
#define HDIM  256

// GEMM block tiling: 256 threads = 8 waves; block tile 128(M) x 32(N);
// each wave owns a 16x32 sub-tile (two 16x16 WMMA accumulators).
#define BM    128
#define BN    32
#define KMAX  HDIM                    // largest K we ever see (256)

typedef __attribute__((ext_vector_type(16))) __bf16 v16bf;
typedef __attribute__((ext_vector_type(8)))  float  v8f;

// -------------------------------------------------------------------------
// Scatter-add: agg[dst[e]] += x[src[e]] over edges, feature dim = Fd (mult of 4).
// One thread handles 4 floats (float4 gather + 4 hw f32 atomics; operands are
// L2-resident so this runs near L2 bandwidth).
// -------------------------------------------------------------------------
__global__ void scatter_add_kernel(const float* __restrict__ x,
                                   const int* __restrict__ src,
                                   const int* __restrict__ dst,
                                   float* __restrict__ agg,
                                   int nE, int Fd)
{
    int chunks = Fd >> 2;
    long long tid = (long long)blockIdx.x * blockDim.x + threadIdx.x;
    long long total = (long long)nE * chunks;
    if (tid >= total) return;
    int e = (int)(tid / chunks);
    int c = (int)(tid % chunks) << 2;
    int s = src[e], d = dst[e];
    float4 v = *(const float4*)(x + (size_t)s * Fd + c);
    float* base = agg + (size_t)d * Fd + c;
    unsafeAtomicAdd(base + 0, v.x);
    unsafeAtomicAdd(base + 1, v.y);
    unsafeAtomicAdd(base + 2, v.z);
    unsafeAtomicAdd(base + 3, v.w);
}

// -------------------------------------------------------------------------
// One-time fp32 -> bf16 conversion of weight matrices (tiny: 128K elements).
// -------------------------------------------------------------------------
__global__ void f32_to_bf16_kernel(const float* __restrict__ in,
                                   __bf16* __restrict__ out, int n)
{
    int i = blockIdx.x * blockDim.x + threadIdx.x;
    if (i < n) out[i] = (__bf16)in[i];
}

// -------------------------------------------------------------------------
// Fused dual GEMM + bias (+relu):
//   C[M,N] = A1[M,K] @ W1[K,N] + A2[M,K] @ W2[K,N] + bias[N]
// Weights arrive pre-converted to bf16. Each block stages its K x 32 weight
// panel (both passes) in LDS, swizzled into the exact per-lane WMMA B-fragment
// order, so the hot loop's B fetch is two contiguous ds_load_b128 per WMMA.
// K % 32 == 0, N % 32 == 0; M handled with row clamping (EXEC stays all-1s).
// -------------------------------------------------------------------------
__global__ __launch_bounds__(256)
void wmma_dual_gemm_kernel(const float* __restrict__ A1,
                           const __bf16* __restrict__ W1b,
                           const float* __restrict__ A2,
                           const __bf16* __restrict__ W2b,
                           const float* __restrict__ bias,
                           float* __restrict__ C,
                           int M, int K, int N, int do_relu)
{
    // 2 passes x (K/32) k-steps x 2 n-subtiles x 1KB fragment-ordered tiles.
    __shared__ __align__(32) char ldsB[2 * (KMAX / 32) * 2 * 1024];

    const int tid      = threadIdx.x;
    const int nsteps   = K >> 5;                    // k-steps of 32
    const int ntBase   = blockIdx.y * BN;           // first output column

    // ---- stage weight panel into LDS in B-fragment order ----
    // B fragment order used here (32x16, same k order as A): lane (h*16+l) holds the 16
    // K-contiguous bf16 values W[k0 + h*16 .. k0 + h*16 + 15][n], n = s*16+l.
    for (int idx = tid; idx < 2 * K * BN; idx += 256) {
        int pass = idx / (K * BN);
        int r    = idx - pass * K * BN;
        int k    = r >> 5;                          // BN == 32
        int nn   = r & 31;
        const __bf16* Wb = pass ? W2b : W1b;
        __bf16 val = Wb[(size_t)k * N + ntBase + nn];
        int t = k >> 5, h = (k >> 4) & 1, j = k & 15;
        int s = nn >> 4, l = nn & 15;
        int lane = h * 16 + l;
        int off = (((pass * nsteps + t) * 2 + s) << 10) + lane * 32 + j * 2;
        *(__bf16*)(ldsB + off) = val;
    }
    __syncthreads();

    const int wave = tid >> 5;
    const int lane = tid & 31;
    const int h    = lane >> 4;                     // lane half
    const int l    = lane & 15;

    // Row this lane loads for the A fragment; clamp so EXEC stays full.
    long long mrow = (long long)blockIdx.x * BM + wave * 16 + l;
    if (mrow > (long long)M - 1) mrow = (long long)M - 1;

    v8f acc0 = {};                                  // cols ntBase + 0..15
    v8f acc1 = {};                                  // cols ntBase + 16..31

    #pragma unroll 1
    for (int pass = 0; pass < 2; ++pass) {
        const float* __restrict__ A = pass ? A2 : A1;
        const float* arowBase = A + (size_t)mrow * K;

        #pragma unroll 1
        for (int t = 0; t < nsteps; ++t) {
            const float* arow = arowBase + t * 32;
            // prefetch next k-slice of this row (global_prefetch_b8)
            __builtin_prefetch(arow + 32, 0, 1);

            // ---- A fragment: 16x32 bf16 in this kernel's own k order ----
            // lane half h: elems 0-15 -> K = h*16 + j (matches B staging)
            float4 a0 = *(const float4*)(arow + h * 16);
            float4 a1 = *(const float4*)(arow + h * 16 + 4);
            float4 a2 = *(const float4*)(arow + h * 16 + 8);
            float4 a3 = *(const float4*)(arow + h * 16 + 12);
            v16bf a;
            a[0]  = (__bf16)a0.x;  a[1]  = (__bf16)a0.y;
            a[2]  = (__bf16)a0.z;  a[3]  = (__bf16)a0.w;
            a[4]  = (__bf16)a1.x;  a[5]  = (__bf16)a1.y;
            a[6]  = (__bf16)a1.z;  a[7]  = (__bf16)a1.w;
            a[8]  = (__bf16)a2.x;  a[9]  = (__bf16)a2.y;
            a[10] = (__bf16)a2.z;  a[11] = (__bf16)a2.w;
            a[12] = (__bf16)a3.x;  a[13] = (__bf16)a3.y;
            a[14] = (__bf16)a3.z;  a[15] = (__bf16)a3.w;

            // ---- B fragments straight from LDS (fragment-ordered) ----
            int boff = ((pass * nsteps + t) * 2) << 10;
            v16bf b0 = *(const v16bf*)(ldsB + boff + lane * 32);
            v16bf b1 = *(const v16bf*)(ldsB + boff + 1024 + lane * 32);

            acc0 = __builtin_amdgcn_wmma_f32_16x16x32_bf16(
                false, a, false, b0, (short)0, acc0, false, false);
            acc1 = __builtin_amdgcn_wmma_f32_16x16x32_bf16(
                false, a, false, b1, (short)0, acc1, false, false);
        }
    }

    // ---- epilogue: C/D f32 16x16 layout: VGPR i -> row (i + 8h), col l ----
    const float bv0 = bias[ntBase + l];
    const float bv1 = bias[ntBase + 16 + l];
    #pragma unroll
    for (int i = 0; i < 8; ++i) {
        long long row = (long long)blockIdx.x * BM + wave * 16 + i + 8 * h;
        if (row < M) {
            float v0 = acc0[i] + bv0;
            float v1 = acc1[i] + bv1;
            if (do_relu) { v0 = fmaxf(v0, 0.0f); v1 = fmaxf(v1, 0.0f); }
            C[(size_t)row * N + ntBase + l]      = v0;
            C[(size_t)row * N + ntBase + 16 + l] = v1;
        }
    }
}

// -------------------------------------------------------------------------
// Pair readout: out[i] = sigmoid(hp2[m0]·w[0:128] + hp2[m1]·w[128:256] + b)
// -------------------------------------------------------------------------
__global__ void readout_kernel(const float* __restrict__ hp2,
                               const float* __restrict__ wlin,
                               const float* __restrict__ blin,
                               const int* __restrict__ mask,
                               float* __restrict__ out, int npair)
{
    int i = blockIdx.x * blockDim.x + threadIdx.x;
    if (i >= npair) return;
    int p1 = mask[2 * i + 0];
    int p2 = mask[2 * i + 1];
    const float4* r1 = (const float4*)(hp2 + (size_t)p1 * FDIM);
    const float4* r2 = (const float4*)(hp2 + (size_t)p2 * FDIM);
    const float4* w1 = (const float4*)(wlin);
    const float4* w2 = (const float4*)(wlin + FDIM);
    float s = blin[0];
    #pragma unroll 4
    for (int j = 0; j < FDIM / 4; ++j) {
        float4 a = r1[j], b = w1[j];
        s += a.x * b.x + a.y * b.y + a.z * b.z + a.w * b.w;
        float4 c = r2[j], d = w2[j];
        s += c.x * d.x + c.y * d.y + c.z * d.z + c.w * d.w;
    }
    out[i] = 1.0f / (1.0f + __expf(-s));
}

// -------------------------------------------------------------------------
extern "C" void kernel_launch(void* const* d_in, const int* in_sizes, int n_in,
                              void* d_out, int out_size, void* d_ws, size_t ws_size,
                              hipStream_t stream)
{
    // Inputs (setup_inputs order). Only the live (ppi + readout) tensors are
    // used: the pos/neg/link paths and hc/hc2 are dead code in the reference.
    const float* x_protein   = (const float*)d_in[0];
    const float* wa_ppi_rel  = (const float*)d_in[20];  // (128,256)
    const float* ba_ppi      = (const float*)d_in[21];  // (256,)
    const float* wa_ppi_root = (const float*)d_in[22];  // (128,256)
    const float* wb_ppi_rel  = (const float*)d_in[23];  // (256,128)
    const float* bb_ppi      = (const float*)d_in[24];  // (128,)
    const float* wb_ppi_root = (const float*)d_in[25];  // (256,128)
    const float* w_lin       = (const float*)d_in[26];  // (256,1)
    const float* b_lin       = (const float*)d_in[27];  // (1,)
    const int*   e_ppi_src   = (const int*)d_in[34];
    const int*   e_ppi_dst   = (const int*)d_in[35];
    const int*   mask        = (const int*)d_in[36];
    float*       out         = (float*)d_out;

    // Workspace layout:
    //   agg1 : NP x 128 f32 (25.6 MB)
    //   hp   : NP x 256 f32 (51.2 MB)
    //   agg2 : NP x 256 f32 (51.2 MB)
    //   hp2  : NP x 128 f32 (25.6 MB)
    //   4 bf16 weight copies (64 KB each)
    char* ws = (char*)d_ws;
    float* agg1 = (float*)(ws);
    float* hp   = (float*)(ws + (size_t)NP * FDIM * sizeof(float));
    float* agg2 = (float*)(ws + (size_t)NP * (FDIM + HDIM) * sizeof(float));
    float* hp2  = (float*)(ws + (size_t)NP * (FDIM + 2 * HDIM) * sizeof(float));
    char*  wb   = ws + (size_t)NP * (2 * FDIM + 2 * HDIM) * sizeof(float);
    __bf16* wa_rel_b  = (__bf16*)(wb);
    __bf16* wa_root_b = (__bf16*)(wb + 1 * 65536);
    __bf16* wb_rel_b  = (__bf16*)(wb + 2 * 65536);
    __bf16* wb_root_b = (__bf16*)(wb + 3 * 65536);

    // Zero the scatter accumulators (graph-capture-safe memset nodes).
    hipMemsetAsync(agg1, 0, (size_t)NP * FDIM * sizeof(float), stream);
    hipMemsetAsync(agg2, 0, (size_t)NP * HDIM * sizeof(float), stream);

    // 0) one-time weight conversion fp32 -> bf16 (4 x 32768 elements)
    {
        int n = FDIM * HDIM;                     // 32768, same for all four
        int blocks = (n + 255) / 256;
        f32_to_bf16_kernel<<<blocks, 256, 0, stream>>>(wa_ppi_rel,  wa_rel_b,  n);
        f32_to_bf16_kernel<<<blocks, 256, 0, stream>>>(wa_ppi_root, wa_root_b, n);
        f32_to_bf16_kernel<<<blocks, 256, 0, stream>>>(wb_ppi_rel,  wb_rel_b,  n);
        f32_to_bf16_kernel<<<blocks, 256, 0, stream>>>(wb_ppi_root, wb_root_b, n);
    }

    // 1) agg1[dst] += x_protein[src]  (F = 128)
    {
        long long total = (long long)EPPI * (FDIM / 4);
        int blocks = (int)((total + 255) / 256);
        scatter_add_kernel<<<blocks, 256, 0, stream>>>(
            x_protein, e_ppi_src, e_ppi_dst, agg1, EPPI, FDIM);
    }

    // 2) hp = relu(agg1 @ wa_ppi_rel + x_protein @ wa_ppi_root + ba_ppi)  [NP x 256]
    {
        dim3 grid((NP + BM - 1) / BM, HDIM / BN);
        wmma_dual_gemm_kernel<<<grid, 256, 0, stream>>>(
            agg1, wa_rel_b, x_protein, wa_root_b, ba_ppi, hp,
            NP, FDIM, HDIM, /*relu=*/1);
    }

    // 3) agg2[dst] += hp[src]  (H = 256)
    {
        long long total = (long long)EPPI * (HDIM / 4);
        int blocks = (int)((total + 255) / 256);
        scatter_add_kernel<<<blocks, 256, 0, stream>>>(
            hp, e_ppi_src, e_ppi_dst, agg2, EPPI, HDIM);
    }

    // 4) hp2 = agg2 @ wb_ppi_rel + hp @ wb_ppi_root + bb_ppi  [NP x 128], no relu
    {
        dim3 grid((NP + BM - 1) / BM, FDIM / BN);
        wmma_dual_gemm_kernel<<<grid, 256, 0, stream>>>(
            agg2, wb_rel_b, hp, wb_root_b, bb_ppi, hp2,
            NP, HDIM, FDIM, /*relu=*/0);
    }

    // 5) pair readout + sigmoid
    {
        int blocks = (NPAIR + 255) / 256;
        readout_kernel<<<blocks, 256, 0, stream>>>(
            hp2, w_lin, b_lin, mask, out, NPAIR);
    }
}
